// GraNNy_ViPeR_48000554500751
// MI455X (gfx1250) — matrix operand order repaired and checked by reading, not compiled, not run on hardware
//
#include <hip/hip_runtime.h>
#include <hip/hip_bf16.h>

typedef __attribute__((ext_vector_type(16))) __bf16 v16bf;
typedef __attribute__((ext_vector_type(8)))  float  v8f;

#define NN 262144
#define NE 2097152

__device__ __forceinline__ void atomAddF(float* p, float v) {
  __hip_atomic_fetch_add(p, v, __ATOMIC_RELAXED, __HIP_MEMORY_SCOPE_AGENT);
}
__device__ __forceinline__ float sigm(float v) {
  return 1.0f / (1.0f + __expf(-v));
}

// ---------------- utility ----------------
__global__ void granny_fill0(float* __restrict__ p, int n) {
  int i = blockIdx.x * 256 + threadIdx.x;
  if (i < n) p[i] = 0.0f;
}

// ---------------- graph prep ----------------
__global__ void granny_degree(const int* __restrict__ dst, float* __restrict__ deg) {
  int e = blockIdx.x * 256 + threadIdx.x;
  if (e < NE) atomAddF(&deg[dst[e]], 1.0f);
}

__global__ void granny_node_prep(const float* __restrict__ deg,
                                 float* __restrict__ invdeg,
                                 float* __restrict__ dinv) {
  int i = blockIdx.x * 256 + threadIdx.x;
  if (i >= NN) return;
  float d = deg[i];
  float dm = fmaxf(d, 1.0f);
  invdeg[i] = 1.0f / dm;
  dinv[i] = (d > 0.0f) ? __frsqrt_rn(dm) : 0.0f;
}

__global__ void granny_edge_norm(const int* __restrict__ src, const int* __restrict__ dst,
                                 const float* __restrict__ dinv, float* __restrict__ norm) {
  int e = blockIdx.x * 256 + threadIdx.x;
  if (e < NE) norm[e] = dinv[src[e]] * dinv[dst[e]];
}

// ---------------- SAGE aggregation ----------------
// F=14 input aggregation: one thread per (edge, feature)
__global__ void granny_sage_agg14(const int* __restrict__ src, const int* __restrict__ dst,
                                  const float* __restrict__ x, float* __restrict__ agg) {
  int gid = blockIdx.x * 256 + threadIdx.x;
  int e = gid >> 4;
  int f = gid & 15;
  if (e >= NE || f >= 14) return;
  atomAddF(&agg[(size_t)dst[e] * 14 + f], x[(size_t)src[e] * 14 + f]);
}

// F=128 aggregation: one thread per (edge, 4-feature chunk), float4 gather + 4 atomics
__global__ void granny_sage_agg128(const int* __restrict__ src, const int* __restrict__ dst,
                                   const float* __restrict__ h, float* __restrict__ agg) {
  int gid = blockIdx.x * 256 + threadIdx.x;
  int e = gid >> 5;
  int fq = (gid & 31) << 2;
  float4 v = *(const float4*)(h + (size_t)src[e] * 128 + fq);
  float* p = agg + (size_t)dst[e] * 128 + fq;
  atomAddF(p + 0, v.x);
  atomAddF(p + 1, v.y);
  atomAddF(p + 2, v.z);
  atomAddF(p + 3, v.w);
}

// ---------------- SAGE WMMA GEMM, middle layers ----------------
// out[n,:] = sigmoid( (agg[n,:]*invdeg[n]) @ Wl + h[n,:] @ Wr + b ), all 128-dim.
// Fused as (N x 256) @ (256 x 128) in bf16 via v_wmma_f32_16x16x32_bf16.
// Block = 256 threads = 8 waves; block covers 128 rows; each wave one 16-row stripe
// across eight 16-col tiles. Weights staged in LDS as bf16 (64 KB).
__global__ __launch_bounds__(256) void granny_sage_gemm_main(
    const float* __restrict__ agg, const float* __restrict__ h,
    const float* __restrict__ invdeg,
    const float* __restrict__ Wl, const float* __restrict__ Wr,
    const float* __restrict__ bias, float* __restrict__ out) {
  __shared__ alignas(32) __bf16 Ws[256 * 128];  // [Ncol=128][K=256] packed [Wl;Wr]
  const int tid = threadIdx.x;
  for (int idx = tid; idx < 256 * 128; idx += 256) {
    int k = idx >> 7, c = idx & 127;
    float w = (k < 128) ? Wl[k * 128 + c] : Wr[(k - 128) * 128 + c];
    Ws[c * 256 + k] = (__bf16)w;
  }
  __syncthreads();

  const int wave = tid >> 5;
  const int lane = tid & 31;
  const int hiHalf = lane >> 4;
  const int m0 = blockIdx.x * 128 + wave * 16;
  const int rowA = m0 + (lane & 15);
  const float scaleA = invdeg[rowA];

  v8f acc[8] = {};

  for (int ks = 0; ks < 8; ++ks) {
    const int k0 = ks * 32;
    v16bf a;
    // A fragment, contiguous K order: lane<16 holds K 0-15,
    // lane>=16 holds K 16-31. Convert f32 -> bf16 inline, folding invdeg.
#pragma unroll
    for (int r = 0; r < 2; ++r) {
      const int kt = k0 + hiHalf * 16 + r * 8;  // run start in concat-K space
      const float* srcp;
      float sc;
      if (kt < 128) { srcp = agg + (size_t)rowA * 128 + kt;        sc = scaleA; }
      else          { srcp = h   + (size_t)rowA * 128 + (kt - 128); sc = 1.0f;  }
      const float4 f0 = *(const float4*)(srcp);
      const float4 f1 = *(const float4*)(srcp + 4);
      a[r * 8 + 0] = (__bf16)(f0.x * sc); a[r * 8 + 1] = (__bf16)(f0.y * sc);
      a[r * 8 + 2] = (__bf16)(f0.z * sc); a[r * 8 + 3] = (__bf16)(f0.w * sc);
      a[r * 8 + 4] = (__bf16)(f1.x * sc); a[r * 8 + 5] = (__bf16)(f1.y * sc);
      a[r * 8 + 6] = (__bf16)(f1.z * sc); a[r * 8 + 7] = (__bf16)(f1.w * sc);
    }
#pragma unroll
    for (int ct = 0; ct < 8; ++ct) {
      // B fragment: lane L holds B[K = k0+16*(L/16)..+15, n0+L%16] (32 contiguous bytes in LDS)
      const v16bf b = *(const v16bf*)&Ws[(size_t)(ct * 16 + (lane & 15)) * 256 + k0 + hiHalf * 16];
      acc[ct] = __builtin_amdgcn_wmma_f32_16x16x32_bf16(
          false, a, false, b, (short)0, acc[ct], false, false);
    }
  }

  // Epilogue: C layout — VGPR r, lane<16 -> (M=r, N=lane); lane>=16 -> (M=r+8, N=lane-16)
  const int col0 = lane & 15;
  const int rbase = m0 + hiHalf * 8;
#pragma unroll
  for (int ct = 0; ct < 8; ++ct) {
#pragma unroll
    for (int r = 0; r < 8; ++r) {
      const int row = rbase + r;
      const int col = ct * 16 + col0;
      out[(size_t)row * 128 + col] = sigm(acc[ct][r] + bias[col]);
    }
  }
}

// ---------------- SAGE WMMA GEMM, input layer (F=14, K padded to 32) ----------------
__global__ __launch_bounds__(256) void granny_sage_gemm_in(
    const float* __restrict__ agg14, const float* __restrict__ x,
    const float* __restrict__ invdeg,
    const float* __restrict__ Wl, const float* __restrict__ Wr,
    const float* __restrict__ bias, float* __restrict__ out) {
  __shared__ alignas(32) __bf16 Ws[32 * 128];  // rows 0-13 Wl, 14-27 Wr, 28-31 zero
  const int tid = threadIdx.x;
  for (int idx = tid; idx < 32 * 128; idx += 256) {
    int k = idx >> 7, c = idx & 127;
    float w = (k < 14) ? Wl[k * 128 + c] : ((k < 28) ? Wr[(k - 14) * 128 + c] : 0.0f);
    Ws[c * 32 + k] = (__bf16)w;
  }
  __syncthreads();

  const int wave = tid >> 5;
  const int lane = tid & 31;
  const int hiHalf = lane >> 4;
  const int m0 = blockIdx.x * 128 + wave * 16;
  const int rowA = m0 + (lane & 15);
  const float sc = invdeg[rowA];

  v16bf a;
#pragma unroll
  for (int i = 0; i < 16; ++i) {
    const int kl = hiHalf * 16 + i;  // K within 0..31
    float val;
    if (kl < 14)      val = agg14[(size_t)rowA * 14 + kl] * sc;
    else if (kl < 28) val = x[(size_t)rowA * 14 + (kl - 14)];
    else              val = 0.0f;
    a[i] = (__bf16)val;
  }

  const int col0 = lane & 15;
  const int rbase = m0 + hiHalf * 8;
#pragma unroll
  for (int ct = 0; ct < 8; ++ct) {
    v8f acc = {};
    const v16bf b = *(const v16bf*)&Ws[(size_t)(ct * 16 + (lane & 15)) * 32 + hiHalf * 16];
    acc = __builtin_amdgcn_wmma_f32_16x16x32_bf16(
        false, a, false, b, (short)0, acc, false, false);
#pragma unroll
    for (int r = 0; r < 8; ++r) {
      const int row = rbase + r;
      const int col = ct * 16 + col0;
      out[(size_t)row * 128 + col] = sigm(acc[r] + bias[col]);
    }
  }
}

// ---------------- SAGE output layer (128 -> 1, relu) ----------------
__global__ void granny_sage_out(const float* __restrict__ agg, const float* __restrict__ h,
                                const float* __restrict__ invdeg,
                                const float* __restrict__ Wl3, const float* __restrict__ Wr3,
                                const float* __restrict__ b3, float* __restrict__ x1) {
  int i = blockIdx.x * 256 + threadIdx.x;
  if (i >= NN) return;
  float s = 0.0f;
  const float sc = invdeg[i];
  const float4* ap = (const float4*)(agg + (size_t)i * 128);
  const float4* hp = (const float4*)(h + (size_t)i * 128);
  const float4* wl = (const float4*)Wl3;
  const float4* wr = (const float4*)Wr3;
#pragma unroll 4
  for (int f = 0; f < 32; ++f) {
    float4 a = ap[f], hh = hp[f], l = wl[f], r = wr[f];
    s += sc * (a.x * l.x + a.y * l.y + a.z * l.z + a.w * l.w);
    s += hh.x * r.x + hh.y * r.y + hh.z * r.z + hh.w * r.w;
  }
  x1[i] = fmaxf(s + b3[0], 0.0f);
}

// ---------------- TAG branch (15-dim, stride padded to 16) ----------------
__global__ void granny_pad14(const float* __restrict__ x, float* __restrict__ t_in) {
  int gid = blockIdx.x * 256 + threadIdx.x;
  int i = gid >> 4, f = gid & 15;
  if (i >= NN) return;
  t_in[(size_t)i * 16 + f] = (f < 14) ? x[(size_t)i * 14 + f] : 0.0f;
}

// acc[i,j] (+)= sum_f cur[i,f] * W[f*Lout + j]
__global__ void granny_tag_mm(const float* __restrict__ cur, const float* __restrict__ W,
                              int F, int Lout, float* __restrict__ acc, int init) {
  int i = blockIdx.x * 256 + threadIdx.x;
  if (i >= NN) return;
  float hbuf[16];
  for (int f = 0; f < F; ++f) hbuf[f] = cur[(size_t)i * 16 + f];
  for (int j = 0; j < Lout; ++j) {
    float s = init ? 0.0f : acc[(size_t)i * 16 + j];
    for (int f = 0; f < F; ++f) s += hbuf[f] * W[f * Lout + j];
    acc[(size_t)i * 16 + j] = s;
  }
}

__global__ void granny_tag_prop(const int* __restrict__ src, const int* __restrict__ dst,
                                const float* __restrict__ norm,
                                const float* __restrict__ cur, float* __restrict__ nxt,
                                int F) {
  int gid = blockIdx.x * 256 + threadIdx.x;
  int e = gid >> 4, f = gid & 15;
  if (e >= NE || f >= F) return;
  float v = norm[e] * cur[(size_t)src[e] * 16 + f];
  atomAddF(&nxt[(size_t)dst[e] * 16 + f], v);
}

__global__ void granny_tag_act(const float* __restrict__ acc, const float* __restrict__ b,
                               int Lout, float* __restrict__ outh) {
  int gid = blockIdx.x * 256 + threadIdx.x;
  int i = gid >> 4, f = gid & 15;
  if (i >= NN) return;
  outh[(size_t)i * 16 + f] = (f < Lout) ? sigm(acc[(size_t)i * 16 + f] + b[f]) : 0.0f;
}

__global__ void granny_tag_act_final(const float* __restrict__ acc, const float* __restrict__ b,
                                     float* __restrict__ x3) {
  int i = blockIdx.x * 256 + threadIdx.x;
  if (i >= NN) return;
  x3[i] = fmaxf(acc[(size_t)i * 16] + b[0], 0.0f);
}

// ---------------- final combine + mask ----------------
__global__ void granny_final(const float* __restrict__ x1, const float* __restrict__ x3,
                             const float* __restrict__ Wlin, const float* __restrict__ blin,
                             const unsigned char* __restrict__ sel, const int* __restrict__ y,
                             float* __restrict__ out) {
  int i = blockIdx.x * 256 + threadIdx.x;
  if (i >= NN) return;
  float v = fmaxf(x1[i] * Wlin[0] + x3[i] * Wlin[1] + blin[0], 0.0f);
  if (sel[i] != 0 && y[i] == 0) v = 0.0f;
  out[i] = v;
}

// ---------------- host orchestration ----------------
static inline void fill0(float* p, int n, hipStream_t s) {
  granny_fill0<<<(n + 255) / 256, 256, 0, s>>>(p, n);
}

extern "C" void kernel_launch(void* const* d_in, const int* in_sizes, int n_in,
                              void* d_out, int out_size, void* d_ws, size_t ws_size,
                              hipStream_t stream) {
  const float* x   = (const float*)d_in[0];
  const int*   ei  = (const int*)d_in[1];
  const int*   y   = (const int*)d_in[2];
  const unsigned char* sel = (const unsigned char*)d_in[3];
  const float* Wl1 = (const float*)d_in[4];
  const float* Wr1 = (const float*)d_in[5];
  const float* b1  = (const float*)d_in[6];
  const float* Wl2 = (const float*)d_in[7];
  const float* Wr2 = (const float*)d_in[8];
  const float* b2  = (const float*)d_in[9];
  const float* Wl3 = (const float*)d_in[10];
  const float* Wr3 = (const float*)d_in[11];
  const float* b3  = (const float*)d_in[12];
  const float* Wt1 = (const float*)d_in[13];
  const float* bt1 = (const float*)d_in[14];
  const float* Wt2 = (const float*)d_in[15];
  const float* bt2 = (const float*)d_in[16];
  const float* Wt3 = (const float*)d_in[17];
  const float* bt3 = (const float*)d_in[18];
  const float* Wlin = (const float*)d_in[19];
  const float* blin = (const float*)d_in[20];
  float* out = (float*)d_out;

  const int* src = ei;
  const int* dst = ei + NE;

  size_t off = 0;
  char* base = (char*)d_ws;
  auto alloc = [&](size_t bytes) -> char* {
    char* p = base + off;
    off += (bytes + 255) & ~(size_t)255;
    return p;
  };
  float* deg    = (float*)alloc((size_t)NN * 4);
  float* invdeg = (float*)alloc((size_t)NN * 4);
  float* dinv   = (float*)alloc((size_t)NN * 4);
  float* norm   = (float*)alloc((size_t)NE * 4);
  float* agg    = (float*)alloc((size_t)NN * 128 * 4);
  float* hA     = (float*)alloc((size_t)NN * 128 * 4);
  float* hB     = (float*)alloc((size_t)NN * 128 * 4);
  float* x1     = (float*)alloc((size_t)NN * 4);
  float* t_in   = (float*)alloc((size_t)NN * 16 * 4);
  float* hop_a  = (float*)alloc((size_t)NN * 16 * 4);
  float* hop_b  = (float*)alloc((size_t)NN * 16 * 4);
  float* t_acc  = (float*)alloc((size_t)NN * 16 * 4);
  float* x3     = (float*)alloc((size_t)NN * 4);
  (void)ws_size; (void)in_sizes; (void)n_in; (void)out_size;

  // --- graph normalization terms ---
  fill0(deg, NN, stream);
  granny_degree<<<NE / 256, 256, 0, stream>>>(dst, deg);
  granny_node_prep<<<NN / 256, 256, 0, stream>>>(deg, invdeg, dinv);
  granny_edge_norm<<<NE / 256, 256, 0, stream>>>(src, dst, dinv, norm);

  // --- SAGE branch ---
  fill0(agg, NN * 14, stream);
  granny_sage_agg14<<<(NE * 16) / 256, 256, 0, stream>>>(src, dst, x, agg);
  granny_sage_gemm_in<<<NN / 128, 256, 0, stream>>>(agg, x, invdeg, Wl1, Wr1, b1, hA);

  float* hc = hA;
  float* hn = hB;
  for (int l = 0; l < 4; ++l) {
    fill0(agg, NN * 128, stream);
    granny_sage_agg128<<<(NE * 32) / 256, 256, 0, stream>>>(src, dst, hc, agg);
    granny_sage_gemm_main<<<NN / 128, 256, 0, stream>>>(agg, hc, invdeg, Wl2, Wr2, b2, hn);
    float* t = hc; hc = hn; hn = t;
  }
  fill0(agg, NN * 128, stream);
  granny_sage_agg128<<<(NE * 32) / 256, 256, 0, stream>>>(src, dst, hc, agg);
  granny_sage_out<<<NN / 256, 256, 0, stream>>>(agg, hc, invdeg, Wl3, Wr3, b3, x1);

  // --- TAG branch ---
  granny_pad14<<<(NN * 16) / 256, 256, 0, stream>>>(x, t_in);

  auto tag_layer = [&](const float* Wt, const float* bt, int F, int Lout, bool final_layer) {
    granny_tag_mm<<<NN / 256, 256, 0, stream>>>(t_in, Wt, F, Lout, t_acc, 1);
    const float* cur = t_in;
    float* bufs[2] = {hop_a, hop_b};
    int pb = 0;
    for (int k = 1; k <= 15; ++k) {
      float* nxt = bufs[pb]; pb ^= 1;
      fill0(nxt, NN * 16, stream);
      granny_tag_prop<<<(NE * 16) / 256, 256, 0, stream>>>(src, dst, norm, cur, nxt, F);
      granny_tag_mm<<<NN / 256, 256, 0, stream>>>(nxt, Wt + (size_t)k * F * Lout, F, Lout, t_acc, 0);
      cur = nxt;
    }
    if (!final_layer)
      granny_tag_act<<<(NN * 16) / 256, 256, 0, stream>>>(t_acc, bt, Lout, t_in);
    else
      granny_tag_act_final<<<NN / 256, 256, 0, stream>>>(t_acc, bt, x3);
  };

  tag_layer(Wt1, bt1, 14, 15, false);
  for (int l = 0; l < 4; ++l) tag_layer(Wt2, bt2, 15, 15, false);
  tag_layer(Wt3, bt3, 15, 1, true);

  // --- combine + mask ---
  granny_final<<<NN / 256, 256, 0, stream>>>(x1, x3, Wlin, blin, sel, y, out);
}
